// LlamaAttention_5523327942918
// MI455X (gfx1250) — matrix operand order repaired and checked by reading, not compiled, not run on hardware
//
#include <hip/hip_runtime.h>

typedef unsigned short u16;
typedef __attribute__((ext_vector_type(16))) __bf16 v16bf;
typedef __attribute__((ext_vector_type(8)))  float  v8f;

#define B_    2
#define S_    2048
#define HID_  4096
#define NH_   32
#define NKV_  8
#define HD_   128
#define SCALING_ 0.08838834764831845f   // 128^-0.5

union Frag { uint4 u2[2]; v16bf v; };

__device__ __forceinline__ u16 f2bf(float f) {
  unsigned u = __float_as_uint(f);
  u += 0x7FFFu + ((u >> 16) & 1u);           // round-to-nearest-even
  return (u16)(u >> 16);
}

// A fragment (16x32 bf16, MxK): lane&15 = row; interleaved K chunks per ISA layout.
__device__ __forceinline__ v16bf loadA(const u16* base, int ld) {
  int lane = threadIdx.x & 31;
  const u16* p = base + (size_t)(lane & 15) * ld + ((lane >> 4) << 3);
  Frag f;
  f.u2[0] = *(const uint4*)p;          // K = kh .. kh+7
  f.u2[1] = *(const uint4*)(p + 16);   // K = kh+16 .. kh+23
  return f.v;
}

// B fragment (32x16 bf16, KxN) from row-major src where row n supplies column n:
// lanes 0-15: K=0..7,16..23; lanes 16-31: K=8..15,24..31 (interleaved, matches A).
__device__ __forceinline__ v16bf loadB(const u16* base, int ld) {
  int lane = threadIdx.x & 31;
  const u16* p = base + (size_t)(lane & 15) * ld + ((lane >> 4) << 3);
  Frag f;
  f.u2[0] = *(const uint4*)p;          // K = kh .. kh+7
  f.u2[1] = *(const uint4*)(p + 16);   // K = kh+16 .. kh+23
  return f.v;
}

__device__ __forceinline__ v8f wmma_bf16(v16bf a, v16bf b, v8f c) {
  return __builtin_amdgcn_wmma_f32_16x16x32_bf16(false, a, false, b, (short)0, c,
                                                 false, false);
}

__device__ __forceinline__ float redmax16(float v) {
  #pragma unroll
  for (int m = 1; m < 16; m <<= 1) v = fmaxf(v, __shfl_xor(v, m, 32));
  return v;
}
__device__ __forceinline__ float redsum16(float v) {
  #pragma unroll
  for (int m = 1; m < 16; m <<= 1) v += __shfl_xor(v, m, 32);
  return v;
}

// ---------------------------------------------------------------------------
// C[M,N] (f32) = A[M,K](bf16, row-major) * W[N,K](bf16, row-major)^T
// 256 threads = 8 waves; wave tile 32x64 (2x4 WMMA => 8 wmma per 6 fragment
// loads, 24 B/wmma per lane); block tile 128x128 (waves 4x2).
// Working set is L2-resident (192MB) so no LDS staging.
// ---------------------------------------------------------------------------
__global__ __launch_bounds__(256) void gemm_bf16_wmma(
    const u16* __restrict__ A, const u16* __restrict__ W,
    float* __restrict__ C, int M, int N, int K)
{
  int lane = threadIdx.x & 31;
  int wave = threadIdx.x >> 5;
  int row0 = blockIdx.y * 128 + (wave >> 1) * 32;   // 4 waves in M
  int col0 = blockIdx.x * 128 + (wave & 1) * 64;    // 2 waves in N

  const u16* A0 = A + (size_t)row0 * K;
  const u16* A1 = A0 + (size_t)16 * K;
  const u16* W0 = W + (size_t)col0 * K;
  const u16* W1 = W0 + (size_t)16 * K;
  const u16* W2 = W0 + (size_t)32 * K;
  const u16* W3 = W0 + (size_t)48 * K;

  v8f c00 = {}, c01 = {}, c02 = {}, c03 = {};
  v8f c10 = {}, c11 = {}, c12 = {}, c13 = {};
  for (int kk = 0; kk < K; kk += 32) {
    v16bf a0 = loadA(A0 + kk, K);
    v16bf a1 = loadA(A1 + kk, K);
    v16bf b0 = loadB(W0 + kk, K);
    v16bf b1 = loadB(W1 + kk, K);
    v16bf b2 = loadB(W2 + kk, K);
    v16bf b3 = loadB(W3 + kk, K);
    c00 = wmma_bf16(a0, b0, c00);
    c01 = wmma_bf16(a0, b1, c01);
    c02 = wmma_bf16(a0, b2, c02);
    c03 = wmma_bf16(a0, b3, c03);
    c10 = wmma_bf16(a1, b0, c10);
    c11 = wmma_bf16(a1, b1, c11);
    c12 = wmma_bf16(a1, b2, c12);
    c13 = wmma_bf16(a1, b3, c13);
  }

  int rbase = (lane >> 4) << 3;
  #pragma unroll
  for (int r = 0; r < 8; ++r) {
    size_t idx0 = (size_t)(row0 + r + rbase) * N + col0 + (lane & 15);
    size_t idx1 = idx0 + (size_t)16 * N;
    C[idx0]      = c00[r];
    C[idx0 + 16] = c01[r];
    C[idx0 + 32] = c02[r];
    C[idx0 + 48] = c03[r];
    C[idx1]      = c10[r];
    C[idx1 + 16] = c11[r];
    C[idx1 + 32] = c12[r];
    C[idx1 + 48] = c13[r];
  }
}

// f32 -> bf16 bulk convert
__global__ void cvt_bf16(const float* __restrict__ in, u16* __restrict__ out, size_t n) {
  size_t i = (size_t)blockIdx.x * blockDim.x + threadIdx.x;
  size_t stride = (size_t)gridDim.x * blockDim.x;
  for (; i < n; i += stride) out[i] = f2bf(in[i]);
}

// RoPE + pack [b,s,h,d] f32 -> [b,h,s,d] bf16
__global__ void rope_pack(const float* __restrict__ x, const float* __restrict__ cs,
                          const float* __restrict__ sn, u16* __restrict__ xb, int nh)
{
  int d = threadIdx.x;                        // 0..127
  int s = blockIdx.x, h = blockIdx.y, b = blockIdx.z;
  size_t src = ((size_t)(b * S_ + s) * nh + h) * HD_;
  float v  = x[src + d];
  float vr = (d < HD_ / 2) ? -x[src + d + HD_ / 2] : x[src + d - HD_ / 2];
  size_t ci = ((size_t)b * S_ + s) * HD_ + d;
  float o = v * cs[ci] + vr * sn[ci];
  xb[((size_t)(b * nh + h) * S_ + s) * HD_ + d] = f2bf(o);
}

// V: [b,s,h,d] f32 -> [b,h,d,s] bf16 (transposed so PV B-fragments are contiguous)
__global__ void v_transpose(const float* __restrict__ v, u16* __restrict__ vt)
{
  int d = threadIdx.x;
  int s = blockIdx.x, h = blockIdx.y, b = blockIdx.z;
  float x = v[((size_t)(b * S_ + s) * NKV_ + h) * HD_ + d];
  vt[((size_t)(b * NKV_ + h) * HD_ + d) * S_ + s] = f2bf(x);
}

// ---------------------------------------------------------------------------
// Flash attention: one wave per 16-row Q tile, 32-key blocks, causal.
// S-tile = Q(16x128) * K^T -> two 16x16 f32 tiles; online softmax with
// per-(reg,half) row stats (xor-shuffle reductions inside 16-lane halves);
// P (16x32) staged through per-wave LDS to become one bf16 A-fragment;
// O accum = 8 x (16x16) f32 WMMA tiles over HD=128.
// ---------------------------------------------------------------------------
__global__ __launch_bounds__(256) void flash_wmma(
    const u16* __restrict__ Qb, const u16* __restrict__ Kb,
    const u16* __restrict__ Vt, u16* __restrict__ Ctx)
{
  __shared__ u16 pbuf[8][16 * 40];            // padded row stride 40 (80B)
  int lane = threadIdx.x & 31;
  int wave = threadIdx.x >> 5;
  int b = blockIdx.z, h = blockIdx.y;
  int hk = h >> 2;                             // N_REP = NH/NKV = 4
  int q0 = (blockIdx.x * 8 + wave) * 16;
  const u16* Qh = Qb + ((size_t)(b * NH_  + h ) * S_) * HD_;
  const u16* Kh = Kb + ((size_t)(b * NKV_ + hk) * S_) * HD_;
  const u16* Vh = Vt + ((size_t)(b * NKV_ + hk) * HD_) * S_;
  int n16   = lane & 15;
  int rbase = (lane >> 4) << 3;

  v16bf qf[4];
  #pragma unroll
  for (int cc = 0; cc < 4; ++cc)
    qf[cc] = loadA(Qh + (size_t)q0 * HD_ + cc * 32, HD_);

  v8f o[8];
  v8f zero = {};
  #pragma unroll
  for (int i = 0; i < 8; ++i) o[i] = zero;
  float mr[8], lr[8];
  #pragma unroll
  for (int r = 0; r < 8; ++r) { mr[r] = -3.0e38f; lr[r] = 0.0f; }

  u16* pb = &pbuf[wave][0];

  for (int k0 = 0; k0 < q0 + 16; k0 += 32) {
    // scores: two 16x16 tiles (keys k0..k0+15, k0+16..k0+31)
    v8f s0 = zero, s1 = zero;
    #pragma unroll
    for (int cc = 0; cc < 4; ++cc) {
      v16bf bk0 = loadB(Kh + (size_t)k0        * HD_ + cc * 32, HD_);
      v16bf bk1 = loadB(Kh + (size_t)(k0 + 16) * HD_ + cc * 32, HD_);
      s0 = wmma_bf16(qf[cc], bk0, s0);
      s1 = wmma_bf16(qf[cc], bk1, s1);
    }

    float sc[8];
    #pragma unroll
    for (int r = 0; r < 8; ++r) {
      int qrow = q0 + r + rbase;
      float x0 = s0[r] * SCALING_;
      float x1 = s1[r] * SCALING_;
      if (k0 + n16      > qrow) x0 = -1.0e9f;  // causal mask
      if (k0 + 16 + n16 > qrow) x1 = -1.0e9f;
      float mx = redmax16(fmaxf(x0, x1));
      float mn = fmaxf(mr[r], mx);
      float scale = __expf(mr[r] - mn);
      mr[r] = mn;
      float p0 = __expf(x0 - mn);
      float p1 = __expf(x1 - mn);
      lr[r] = lr[r] * scale + redsum16(p0 + p1);
      sc[r] = scale;
      pb[(r + rbase) * 40 + n16]      = f2bf(p0);
      pb[(r + rbase) * 40 + 16 + n16] = f2bf(p1);
    }
    #pragma unroll
    for (int dc = 0; dc < 8; ++dc)
      #pragma unroll
      for (int r = 0; r < 8; ++r) o[dc][r] *= sc[r];

    // per-wave LDS fence (no block barrier: waves have divergent trip counts;
    // DS ops from one wave complete in-order, just keep compiler honest + wait)
    __builtin_amdgcn_wave_barrier();
    asm volatile("s_wait_dscnt 0" ::: "memory");

    v16bf pf = loadA(pb, 40);                  // P as 16x32 bf16 A-fragment
    #pragma unroll
    for (int dc = 0; dc < 8; ++dc) {
      v16bf vfr = loadB(Vh + (size_t)(dc * 16) * S_ + k0, S_);
      o[dc] = wmma_bf16(pf, vfr, o[dc]);
    }
    __builtin_amdgcn_wave_barrier();           // WAR on pbuf for next iter
  }

  float inv[8];
  #pragma unroll
  for (int r = 0; r < 8; ++r) inv[r] = 1.0f / lr[r];
  #pragma unroll
  for (int dc = 0; dc < 8; ++dc)
    #pragma unroll
    for (int r = 0; r < 8; ++r) {
      size_t idx = ((size_t)b * S_ + q0 + r + rbase) * ((size_t)NH_ * HD_)
                 + (size_t)h * HD_ + dc * 16 + n16;
      Ctx[idx] = f2bf(o[dc][r] * inv[r]);
    }
}

// ---------------------------------------------------------------------------
extern "C" void kernel_launch(void* const* d_in, const int* in_sizes, int n_in,
                              void* d_out, int out_size, void* d_ws, size_t ws_size,
                              hipStream_t stream)
{
  (void)in_sizes; (void)n_in; (void)out_size; (void)ws_size;
  const float* hs = (const float*)d_in[0];
  const float* cs = (const float*)d_in[1];
  const float* sn = (const float*)d_in[2];
  // d_in[3] (attention_mask) is the causal mask; applied analytically in-kernel.
  const float* Wq = (const float*)d_in[4];
  const float* Wk = (const float*)d_in[5];
  const float* Wv = (const float*)d_in[6];
  const float* Wo = (const float*)d_in[7];
  float* out = (float*)d_out;

  const size_t M = (size_t)B_ * S_;            // 4096 rows
  char* ws = (char*)d_ws;
  size_t off = 0;
  auto take = [&](size_t bytes) -> char* {
    char* p = ws + off;
    off += (bytes + 255) & ~(size_t)255;
    return p;
  };
  u16*   hs_b = (u16*)  take(M * HID_ * 2);
  u16*   wq_b = (u16*)  take((size_t)NH_  * HD_ * HID_ * 2);
  u16*   wk_b = (u16*)  take((size_t)NKV_ * HD_ * HID_ * 2);
  u16*   wv_b = (u16*)  take((size_t)NKV_ * HD_ * HID_ * 2);
  u16*   wo_b = (u16*)  take((size_t)HID_ * NH_ * HD_ * 2);
  float* qf   = (float*)take(M * NH_  * HD_ * 4);
  float* kf   = (float*)take(M * NKV_ * HD_ * 4);
  float* vf   = (float*)take(M * NKV_ * HD_ * 4);
  u16*   qb   = (u16*)  take(M * NH_  * HD_ * 2);
  u16*   kb   = (u16*)  take(M * NKV_ * HD_ * 2);
  u16*   vt   = (u16*)  take(M * NKV_ * HD_ * 2);
  u16*   ctx  = (u16*)qf;   // alias: qf is dead after rope_pack (stream-ordered)

  cvt_bf16<<<2048, 256, 0, stream>>>(hs, hs_b, M * HID_);
  cvt_bf16<<<2048, 256, 0, stream>>>(Wq, wq_b, (size_t)NH_  * HD_ * HID_);
  cvt_bf16<<<1024, 256, 0, stream>>>(Wk, wk_b, (size_t)NKV_ * HD_ * HID_);
  cvt_bf16<<<1024, 256, 0, stream>>>(Wv, wv_b, (size_t)NKV_ * HD_ * HID_);
  cvt_bf16<<<2048, 256, 0, stream>>>(Wo, wo_b, (size_t)HID_ * NH_ * HD_);

  gemm_bf16_wmma<<<dim3((NH_  * HD_) / 128, (int)M / 128), 256, 0, stream>>>(
      hs_b, wq_b, qf, (int)M, NH_ * HD_, HID_);
  gemm_bf16_wmma<<<dim3((NKV_ * HD_) / 128, (int)M / 128), 256, 0, stream>>>(
      hs_b, wk_b, kf, (int)M, NKV_ * HD_, HID_);
  gemm_bf16_wmma<<<dim3((NKV_ * HD_) / 128, (int)M / 128), 256, 0, stream>>>(
      hs_b, wv_b, vf, (int)M, NKV_ * HD_, HID_);

  rope_pack<<<dim3(S_, NH_,  B_), 128, 0, stream>>>(qf, cs, sn, qb, NH_);
  rope_pack<<<dim3(S_, NKV_, B_), 128, 0, stream>>>(kf, cs, sn, kb, NKV_);
  v_transpose<<<dim3(S_, NKV_, B_), 128, 0, stream>>>(vf, vt);

  flash_wmma<<<dim3(S_ / 128, NH_, B_), 256, 0, stream>>>(qb, kb, vt, ctx);

  gemm_bf16_wmma<<<dim3(HID_ / 128, (int)M / 128), 256, 0, stream>>>(
      ctx, wo_b, out, (int)M, HID_, HID_);
}
